// SensorGAT_4131758539435
// MI455X (gfx1250) — matrix operand order repaired and checked by reading, not compiled, not run on hardware
//
#include <hip/hip_runtime.h>
#include <hip/hip_bf16.h>
#include <float.h>

typedef __attribute__((ext_vector_type(16))) __bf16 v16bf;
typedef __attribute__((ext_vector_type(8)))  float  v8f;

// ---------- helpers ----------
__device__ __forceinline__ __bf16 f2bf(float f) {
  // round-to-nearest-even f32 -> bf16
  unsigned u = __float_as_uint(f);
  unsigned r = u + 0x7FFFu + ((u >> 16) & 1u);
  unsigned short h = (unsigned short)(r >> 16);
  return __builtin_bit_cast(__bf16, h);
}

__device__ __forceinline__ void atomicMaxF(float* addr, float v) {
  // classic monotonic-bit-pattern trick; addr initialized to -FLT_MAX
  if (v >= 0.0f) atomicMax((int*)addr, __float_as_int(v));
  else           atomicMin((unsigned int*)addr, __float_as_uint(v));
}

// ---------- generic float fill ----------
__global__ void k_fill(float* __restrict__ p, float v, int n) {
  int i = blockIdx.x * blockDim.x + threadIdx.x;
  if (i < n) p[i] = v;
}

// ---------- WMMA GEMM: D[nrows x NOUT] = A[nrows x 128] * W[128 x NOUT] ----------
// bf16 inputs (converted in-flight), f32 accumulate via v_wmma_f32_16x16x32_bf16.
template <int NOUT>
__global__ __launch_bounds__(256) void k_gemm_wmma(
    const float* __restrict__ A, const float* __restrict__ W,
    float* __restrict__ D, int nrows) {
  constexpr int K   = 128;
  constexpr int LDW = K + 8;                 // pad: 272B row stride -> conflict-free
  __shared__ __bf16 Wt[NOUT * LDW];          // transposed W: Wt[n][k]

  for (int i = threadIdx.x; i < K * NOUT; i += 256) {
    int k = i / NOUT, n = i - k * NOUT;      // W is row-major [K][NOUT]
    Wt[n * LDW + k] = f2bf(W[i]);
  }
  __syncthreads();

  const int wave = threadIdx.x >> 5;
  const int lane = threadIdx.x & 31;
  const int half = lane >> 4;                // 0: lanes 0-15, 1: lanes 16-31
  const int l16  = lane & 15;
  const int m0   = (blockIdx.x * 8 + wave) * 16;
  if (m0 >= nrows) return;                   // wave-uniform: EXEC stays all-ones

  const int row = (m0 + l16 < nrows) ? (m0 + l16) : (nrows - 1);
  const float* ar = A + (size_t)row * K;

  // A fragment layout (16-bit A 16x32): VGPR0-3 = K in [half*8, half*8+8),
  // VGPR4-7 = K in [16+half*8, 16+half*8+8)
  v16bf afrag[4];
#pragma unroll
  for (int t = 0; t < 4; ++t) {
    const int kb = t * 32 + half * 8;
#pragma unroll
    for (int j = 0; j < 8; ++j) {
      afrag[t][j]     = f2bf(ar[kb + j]);
      afrag[t][j + 8] = f2bf(ar[kb + 16 + j]);
    }
  }

#pragma unroll
  for (int n0 = 0; n0 < NOUT; n0 += 16) {
    const __bf16* wc = &Wt[(n0 + l16) * LDW];
    v8f acc = {};
#pragma unroll
    for (int t = 0; t < 4; ++t) {
      // B fragment (32x16): elements 0-7 hold K=kt+half*8..+7, elements 8-15 hold K=kt+16+half*8..+7
      const __bf16* wk = wc + t * 32 + half * 8;
      v16bf bfrag;
#pragma unroll
      for (int j = 0; j < 8; ++j) { bfrag[j] = wk[j]; bfrag[j + 8] = wk[16 + j]; }
      acc = __builtin_amdgcn_wmma_f32_16x16x32_bf16(
          false, afrag[t], false, bfrag, (short)0, acc, false, false);
    }
    // C/D layout: VGPR r -> M = r + 8*half, N = l16
#pragma unroll
    for (int r = 0; r < 8; ++r) {
      int mr = m0 + r + 8 * half;
      if (mr < nrows) D[(size_t)mr * NOUT + (n0 + l16)] = acc[r];
    }
  }
}

// ---------- per-node attention scores: als/ald [N*H] ----------
__global__ void k_scores(const float* __restrict__ h,
                         const float* __restrict__ a_src,
                         const float* __restrict__ a_dst,
                         float* __restrict__ als, float* __restrict__ ald,
                         int N, int H, int C) {
  int t = blockIdx.x * blockDim.x + threadIdx.x;
  if (t >= N * H) return;
  int n = t / H, hd = t - n * H;
  const float* hp = h + (size_t)n * H * C + hd * C;
  const float* as = a_src + hd * C;
  const float* ad = a_dst + hd * C;
  float s1 = 0.f, s2 = 0.f;
  for (int c = 0; c < C; ++c) { float v = hp[c]; s1 += v * as[c]; s2 += v * ad[c]; }
  als[t] = s1; ald[t] = s2;
}

// ---------- edge pass 1: e = leaky_relu(als[src]+ald[dst]); segment max ----------
__global__ void k_edge_max(const int* __restrict__ src, const int* __restrict__ dst,
                           const float* __restrict__ als, const float* __restrict__ ald,
                           float* __restrict__ ebuf, float* __restrict__ m,
                           int E, int Etot, int H, float slope) {
  int t = blockIdx.x * blockDim.x + threadIdx.x;
  if (t >= Etot * H) return;
  int i = t / H, hd = t - i * H;
  int s, d;
  if (i < E) { s = src[i]; d = dst[i]; } else { s = d = i - E; }  // self-loop tail
  float e = als[s * H + hd] + ald[d * H + hd];
  e = (e > 0.f) ? e : slope * e;
  ebuf[t] = e;
  atomicMaxF(&m[d * H + hd], e);
}

// ---------- edge pass 2: ex = exp(e - m[dst]); segment sum ----------
__global__ void k_edge_sum(const int* __restrict__ src, const int* __restrict__ dst,
                           const float* __restrict__ m, float* __restrict__ ebuf,
                           float* __restrict__ ssum, int E, int Etot, int H) {
  int t = blockIdx.x * blockDim.x + threadIdx.x;
  if (t >= Etot * H) return;
  int i = t / H, hd = t - i * H;
  int d = (i < E) ? dst[i] : (i - E);
  float ex = __expf(ebuf[t] - m[d * H + hd]);
  ebuf[t] = ex;
  atomicAdd(&ssum[d * H + hd], ex);
}

// ---------- edge pass 3: out[dst] += h[src] * alpha ----------
__global__ void k_edge_aggr(const int* __restrict__ src, const int* __restrict__ dst,
                            const float* __restrict__ h, const float* __restrict__ ebuf,
                            const float* __restrict__ ssum, float* __restrict__ out,
                            int E, int Etot, int H, int C) {
  int t = blockIdx.x * blockDim.x + threadIdx.x;
  int HC = H * C;
  if (t >= Etot * HC) return;
  int i = t / HC, c = t - i * HC, hd = c / C;
  int s, d;
  if (i < E) { s = src[i]; d = dst[i]; } else { s = d = i - E; }
  float alpha = ebuf[i * H + hd] / ssum[d * H + hd];
  atomicAdd(&out[(size_t)d * HC + c], h[(size_t)s * HC + c] * alpha);
}

// ---------- bias + ELU (in place) ----------
__global__ void k_bias_elu(float* __restrict__ out, const float* __restrict__ b,
                           int N, int F) {
  int t = blockIdx.x * blockDim.x + threadIdx.x;
  if (t >= N * F) return;
  float v = out[t] + b[t % F];
  out[t] = (v > 0.f) ? v : (__expf(v) - 1.f);
}

// ---------- pooling: pooled[c] = sum_n h2[n][c] (per-thread accumulate + atomicAdd) ----------
__global__ void k_pool(const float* __restrict__ h2, float* __restrict__ pooled,
                       int N, int C) {
  int t = blockIdx.x * blockDim.x + threadIdx.x;
  int c = t & 31;                 // C == 32
  int r0 = t >> 5;
  int rstride = (gridDim.x * blockDim.x) >> 5;
  float acc = 0.f;
  for (int r = r0; r < N; r += rstride) acc += h2[(size_t)r * C + c];
  atomicAdd(&pooled[c], acc);
}

// ---------- final head: softmax((pooled/N + b2) @ linW + linb) ----------
__global__ void k_final(const float* __restrict__ pooled, const float* __restrict__ b2,
                        const float* __restrict__ linW, const float* __restrict__ linb,
                        float* __restrict__ out, float invN, int C, int ncls) {
  if (threadIdx.x != 0 || blockIdx.x != 0) return;
  float logits[8];
  float mx = -FLT_MAX;
  for (int j = 0; j < ncls; ++j) {
    float acc = linb[j];
    for (int c = 0; c < C; ++c) acc += (pooled[c] * invN + b2[c]) * linW[c * ncls + j];
    logits[j] = acc;
    if (acc > mx) mx = acc;
  }
  float sum = 0.f;
  for (int j = 0; j < ncls; ++j) { logits[j] = __expf(logits[j] - mx); sum += logits[j]; }
  for (int j = 0; j < ncls; ++j) out[j] = logits[j] / sum;
}

// ============================ launch ============================
extern "C" void kernel_launch(void* const* d_in, const int* in_sizes, int n_in,
                              void* d_out, int out_size, void* d_ws, size_t ws_size,
                              hipStream_t stream) {
  const float* x    = (const float*)d_in[0];
  const int*   ei   = (const int*)d_in[1];
  const float* W1   = (const float*)d_in[2];
  const float* a1s  = (const float*)d_in[3];
  const float* a1d  = (const float*)d_in[4];
  const float* b1   = (const float*)d_in[5];
  const float* W2   = (const float*)d_in[6];
  const float* a2s  = (const float*)d_in[7];
  const float* a2d  = (const float*)d_in[8];
  const float* b2   = (const float*)d_in[9];
  const float* linW = (const float*)d_in[10];
  const float* linb = (const float*)d_in[11];

  const int FIN = 128, H1 = 4, C1 = 32, HC = 128, C2 = 32, NCLS = 3;
  const int N    = in_sizes[0] / FIN;        // 100000
  const int E    = in_sizes[1] / 2;          // 1600000
  const int Etot = E + N;                    // + self loops
  const int* srcp = ei;
  const int* dstp = ei + E;

  // ---- workspace layout (floats); layer-2 reuses layer-1 regions ----
  float* ws   = (float*)d_ws;
  float* h1   = ws;                          // N*128
  float* als1 = h1   + (size_t)N * HC;       // N*4
  float* ald1 = als1 + (size_t)N * H1;
  float* m1   = ald1 + (size_t)N * H1;
  float* s1   = m1   + (size_t)N * H1;
  float* e1   = s1   + (size_t)N * H1;       // Etot*4
  float* out1 = e1   + (size_t)Etot * H1;    // N*128 (becomes hA = elu(out1+b1))
  float* pooled = out1 + (size_t)N * HC;     // 32
  // layer-2 aliases (layer-1 scratch is dead by then)
  float* h2   = h1;                          // N*32
  float* als2 = als1;
  float* ald2 = ald1;
  float* m2   = m1;
  float* s2   = s1;
  float* e2   = e1;                          // Etot
  float* out2 = e1 + (size_t)Etot;           // N*32 (fits inside old e1 region)

  const int T = 256;
  auto blk = [&](int n) { return (n + T - 1) / T; };

  // ================= layer 1 =================
  k_fill<<<blk(N * H1), T, 0, stream>>>(m1, -FLT_MAX, N * H1);
  k_fill<<<blk(N * H1), T, 0, stream>>>(s1, 0.f, N * H1);
  k_fill<<<blk(N * HC), T, 0, stream>>>(out1, 0.f, N * HC);

  {
    int strips = (N + 15) / 16;
    k_gemm_wmma<128><<<(strips + 7) / 8, 256, 0, stream>>>(x, W1, h1, N);
  }
  k_scores<<<blk(N * H1), T, 0, stream>>>(h1, a1s, a1d, als1, ald1, N, H1, C1);
  k_edge_max<<<blk(Etot * H1), T, 0, stream>>>(srcp, dstp, als1, ald1, e1, m1,
                                               E, Etot, H1, 0.2f);
  k_edge_sum<<<blk(Etot * H1), T, 0, stream>>>(srcp, dstp, m1, e1, s1, E, Etot, H1);
  k_edge_aggr<<<blk(Etot * H1 * C1), T, 0, stream>>>(srcp, dstp, h1, e1, s1, out1,
                                                     E, Etot, H1, C1);
  k_bias_elu<<<blk(N * HC), T, 0, stream>>>(out1, b1, N, HC);

  // ================= layer 2 =================
  k_fill<<<blk(N), T, 0, stream>>>(m2, -FLT_MAX, N);
  k_fill<<<blk(N), T, 0, stream>>>(s2, 0.f, N);
  k_fill<<<blk(N * C2), T, 0, stream>>>(out2, 0.f, N * C2);
  k_fill<<<1, T, 0, stream>>>(pooled, 0.f, C2);

  {
    int strips = (N + 15) / 16;
    k_gemm_wmma<32><<<(strips + 7) / 8, 256, 0, stream>>>(out1, W2, h2, N);
  }
  k_scores<<<blk(N), T, 0, stream>>>(h2, a2s, a2d, als2, ald2, N, 1, C2);
  k_edge_max<<<blk(Etot), T, 0, stream>>>(srcp, dstp, als2, ald2, e2, m2,
                                          E, Etot, 1, 0.2f);
  k_edge_sum<<<blk(Etot), T, 0, stream>>>(srcp, dstp, m2, e2, s2, E, Etot, 1);
  k_edge_aggr<<<blk(Etot * C2), T, 0, stream>>>(srcp, dstp, h2, e2, s2, out2,
                                                E, Etot, 1, C2);

  // ================= readout =================
  k_pool<<<512, 256, 0, stream>>>(out2, pooled, N, C2);
  k_final<<<1, 64, 0, stream>>>(pooled, b2, linW, linb, (float*)d_out,
                                1.0f / (float)N, C2, NCLS);
}
